// EmbeddingModel_45999099740498
// MI455X (gfx1250) — matrix operand order repaired and checked by reading, not compiled, not run on hardware
//
#include <hip/hip_runtime.h>

typedef __attribute__((ext_vector_type(16))) _Float16 v16h;
typedef __attribute__((ext_vector_type(8)))  float    v8f;
typedef __attribute__((ext_vector_type(4)))  float    v4f;

#define VOCAB 100000
#define EMBED 128
#define BATCH 32768
#define NPOS  10
#define NNEG  50

// Convert 8 consecutive floats at p (16B-aligned) into a[base..base+7] as f16.
__device__ __forceinline__ void cvt8(const float* __restrict__ p, v16h& a, int base) {
    v4f x = *(const v4f*)p;
    v4f y = *(const v4f*)(p + 4);
    a[base + 0] = (_Float16)x[0];
    a[base + 1] = (_Float16)x[1];
    a[base + 2] = (_Float16)x[2];
    a[base + 3] = (_Float16)x[3];
    a[base + 4] = (_Float16)y[0];
    a[base + 5] = (_Float16)y[1];
    a[base + 6] = (_Float16)y[2];
    a[base + 7] = (_Float16)y[3];
}

__global__ void __launch_bounds__(256)
w2v_neg_loss_wmma(const float* __restrict__ in_embed,
                  const float* __restrict__ out_embed,
                  const int*   __restrict__ input_labels,
                  const int*   __restrict__ pos_labels,
                  const int*   __restrict__ neg_labels,
                  float*       __restrict__ out)
{
    const int lane = threadIdx.x & 31;          // wave32
    const int wave = blockIdx.x * (blockDim.x >> 5) + (threadIdx.x >> 5);
    if (wave >= BATCH) return;                  // grid covers BATCH exactly (wave-uniform)
    const int b    = wave;
    const int half = lane >> 4;                 // 0: lanes 0-15, 1: lanes 16-31
    const int m    = lane & 15;

    // ---- B matrix: center vector inp[b], replicated across 16 columns ----
    // Packed element i of lane L must be inp[k0 + 16*half + i] for chunk k0.
    const float* inp = in_embed + (size_t)input_labels[b] * EMBED;
    v16h bchunk[4];
    #pragma unroll
    for (int c = 0; c < 4; ++c) {
        const float* src = inp + c * 32 + half * 16;
        cvt8(src,     bchunk[c], 0);
        cvt8(src + 8, bchunk[c], 8);
    }

    // ---- Gather the 64 (padded) context-row base pointers for this lane's A rows ----
    const float* arow[4];
    #pragma unroll
    for (int t = 0; t < 4; ++t) {
        int j = t * 16 + m;
        int lbl;
        if (j < NPOS)             lbl = pos_labels[b * NPOS + j];
        else if (j < NPOS + NNEG) lbl = neg_labels[b * NNEG + (j - NPOS)];
        else                      lbl = 0;      // pad row; masked out of the sum
        arow[t] = out_embed + (size_t)lbl * EMBED;
    }

    // ---- 4 row-tiles x 4 K-chunks of v_wmma_f32_16x16x32_f16 ----
    // A layout: lane L holds row m=L%16; packed elem i -> K = k0+16*half+i (matches B).
    v8f acc[4];
    #pragma unroll
    for (int t = 0; t < 4; ++t) {
        v8f c = {};
        #pragma unroll
        for (int kc = 0; kc < 4; ++kc) {
            const float* src = arow[t] + kc * 32 + half * 16;
            v16h a;
            cvt8(src,      a, 0);
            cvt8(src + 8,  a, 8);
            c = __builtin_amdgcn_wmma_f32_16x16x32_f16(
                    /*neg_a=*/false, a, /*neg_b=*/false, bchunk[kc],
                    /*c_mod=*/(short)0, c, /*reuse_a=*/false, /*reuse_b=*/false);
        }
        acc[t] = c;
    }

    // ---- Accumulator layout: lane holds column n=m, rows r + 8*half.
    // Lane's dot j = t*16 + 8*half + r  (replicated across the 16 lanes of its half).
    float partial = 0.f;
    #pragma unroll
    for (int t = 0; t < 4; ++t) {
        #pragma unroll
        for (int r = 0; r < 8; ++r) {
            int j = t * 16 + half * 8 + r;
            if (j < NPOS + NNEG) {
                float d  = acc[t][r];
                float x  = (j < NPOS) ? d : -d;      // negatives: log_sigmoid(-dot)
                float ax = fabsf(x);
                // numerically stable: log_sigmoid(x) = min(x,0) - log1p(exp(-|x|))
                float ls = fminf(x, 0.f) - __logf(1.f + __expf(-ax));
                partial += ls;
            }
        }
    }
    // halves hold disjoint row sets (each replicated 16x within the half)
    partial += __shfl_xor(partial, 16, 32);
    if (lane == 0) out[b] = -partial;
}

extern "C" void kernel_launch(void* const* d_in, const int* in_sizes, int n_in,
                              void* d_out, int out_size, void* d_ws, size_t ws_size,
                              hipStream_t stream) {
    const float* in_embed     = (const float*)d_in[0];
    const float* out_embed    = (const float*)d_in[1];
    const int*   input_labels = (const int*)  d_in[2];
    const int*   pos_labels   = (const int*)  d_in[3];
    const int*   neg_labels   = (const int*)  d_in[4];
    float*       out          = (float*)      d_out;

    // one wave32 per batch element; 8 waves per 256-thread block
    dim3 grid(BATCH / 8), block(256);
    w2v_neg_loss_wmma<<<grid, block, 0, stream>>>(
        in_embed, out_embed, input_labels, pos_labels, neg_labels, out);
}
